// WeightedStateLoss4_46995532153317
// MI455X (gfx1250) — compile-verified, matrix operand order checked by reading, not run on hardware
//
#include <hip/hip_runtime.h>
#include <hip/hip_bf16.h>

typedef __attribute__((ext_vector_type(16))) _Float16 v16h;
typedef __attribute__((ext_vector_type(8)))  float    v8f;

#define Bq    512
#define Hq    2048
#define Dq    64
#define WAVES 8      // waves per block (wave32)
#define CHUNK 32     // K width of one WMMA step

// Kernel 1: per-batch nonzero count via V_WMMA_F32_16X16X32_F16, then gather+weight.
// Grid: Bq/16 blocks x 256 threads. Each wave handles 16 batch rows x (Hq/WAVES) h's.
__global__ __launch_bounds__(256)
void wsl4_count_gather(const float* __restrict__ pred,
                       const float* __restrict__ targ,
                       float* __restrict__ row_out)
{
    __shared__ float lcnt[WAVES * 16];

    const int tid  = threadIdx.x;
    const int wave = tid >> 5;
    const int lane = tid & 31;
    const int m    = lane & 15;     // matrix row M this lane feeds
    const int hsel = lane >> 4;     // which K-half of A this lane holds
    const int b0   = blockIdx.x * 16;
    const int i    = b0 + m;        // batch row

    const int hspan = Hq / WAVES;   // 256
    const int hbeg  = wave * hspan;

    // A-matrix (16-bit, 16x32) K offsets for this lane's 16 elements:
    // elements 0..7  -> K = e      + 8*hsel
    // elements 8..15 -> K = e + 8  + 8*hsel
    // i.e. koff[e] = e + (e & 8) + 8*hsel
    const float* tp = targ + ((size_t)i * Hq) * Dq + 1;   // channel 1 of row i

    // All-ones B: every column of D becomes the row count (layout-agnostic).
    v16h bmat;
    #pragma unroll
    for (int e = 0; e < 16; ++e) bmat[e] = (_Float16)1.0f;

    v8f c = {0.f, 0.f, 0.f, 0.f, 0.f, 0.f, 0.f, 0.f};

    for (int h0 = hbeg; h0 < hbeg + hspan; h0 += CHUNK) {
        v16h a;
        #pragma unroll
        for (int e = 0; e < 16; ++e) {
            const int k = e + (e & 8) + 8 * hsel;
            float v = __builtin_nontemporal_load(tp + (size_t)(h0 + k) * Dq);
            a[e] = (v != 0.0f) ? (_Float16)1.0f : (_Float16)0.0f;
        }
        // counts += A * ones  (exact: 0/1 in f16, f32 accumulate, t <= 2048 << 2^24)
        c = __builtin_amdgcn_wmma_f32_16x16x32_f16(false, a, false, bmat,
                                                   (short)0, c, false, false);
    }

    // C/D layout: VGPR r, lanes 0-15 -> (M=r, N=lane); lanes 16-31 -> (M=8+r, N=lane-16).
    // Column N=0: lane 0 holds rows 0..7, lane 16 holds rows 8..15.
    if (lane == 0) {
        #pragma unroll
        for (int r = 0; r < 8; ++r) lcnt[wave * 16 + r] = c[r];
    } else if (lane == 16) {
        #pragma unroll
        for (int r = 0; r < 8; ++r) lcnt[wave * 16 + 8 + r] = c[r];
    }
    __syncthreads();

    if (tid < 16) {
        float t = 0.f;
        #pragma unroll
        for (int w = 0; w < WAVES; ++w) t += lcnt[w * 16 + tid];   // exact integer sums

        int ti  = (int)(t + 0.5f);
        int idx = ti - 1;
        if (idx < 0) idx += Hq;          // numpy/JAX negative-index wrap for t==0

        const int    ib  = b0 + tid;
        const size_t off = ((size_t)ib * Hq + (size_t)idx) * Dq;   // channel 0
        const float  d   = pred[off] - targ[off];
        const float  x   = t * (1.0f / (float)(Hq - 1));
        const float  w   = 1.0f + 0.7f * powf(x, 2.5f);
        row_out[ib] = d * d * w * (float)Dq;
    }
}

// Kernel 2: deterministic fixed-order tree reduction of 512 row values -> mean.
__global__ __launch_bounds__(256)
void wsl4_reduce(const float* __restrict__ row, float* __restrict__ out)
{
    __shared__ float s[256];
    const int tid = threadIdx.x;
    s[tid] = row[tid] + row[tid + 256];
    __syncthreads();
    #pragma unroll
    for (int w = 128; w > 0; w >>= 1) {
        if (tid < w) s[tid] += s[tid + w];
        __syncthreads();
    }
    if (tid == 0) out[0] = s[0] * (1.0f / (float)Bq);
}

extern "C" void kernel_launch(void* const* d_in, const int* in_sizes, int n_in,
                              void* d_out, int out_size, void* d_ws, size_t ws_size,
                              hipStream_t stream)
{
    const float* pred = (const float*)d_in[0];
    const float* targ = (const float*)d_in[1];
    float* row = (float*)d_ws;            // 512 floats of scratch
    float* out = (float*)d_out;

    wsl4_count_gather<<<Bq / 16, 256, 0, stream>>>(pred, targ, row);
    wsl4_reduce<<<1, 256, 0, stream>>>(row, out);
}
